// MultiHeadAttention_56581899157504
// MI455X (gfx1250) — matrix operand order repaired and checked by reading, not compiled, not run on hardware
//
#include <hip/hip_runtime.h>
#include <hip/hip_bf16.h>
#include <stdint.h>

typedef __bf16 bf16_t;
typedef __attribute__((ext_vector_type(16))) __bf16 v16bf;
typedef __attribute__((ext_vector_type(8)))  float  v8f;

#define N_HEADC 16
#define D_MODELC 1024
#define D_KC 64
#define BATCHC 4
#define SEQC 2048
#define ROWSC (BATCHC * SEQC) /* 8192 */
#define LN_EPSC 1e-6f

// ---------------- CDNA5 async copy helpers ----------------
// Copies 16B from global directly into LDS, tracked by ASYNCcnt.
__device__ __forceinline__ void async_b128(void* lds, const void* gaddr) {
  uint32_t l = (uint32_t)(uintptr_t)lds;  // LDS aperture: low 32 bits = LDS byte addr
  asm volatile("global_load_async_to_lds_b128 %0, %1, off"
               :: "v"(l), "v"(gaddr)
               : "memory");
}
__device__ __forceinline__ void wait_async0() {
  asm volatile("s_wait_asynccnt 0x0" ::: "memory");
}

// ---------------- WMMA fragment helpers (CDNA5 wave32 layouts) ----------------

// A operand: 16x32 bf16, row-major source with leading dim lda.
__device__ __forceinline__ v16bf load_a_frag(const bf16_t* base, int lda) {
  const int lane = threadIdx.x & 31;
  const int row  = lane & 15;
  const int half = lane >> 4;
  const bf16_t* rp = base + row * lda;
  v16bf a;
#pragma unroll
  for (int i = 0; i < 8; ++i) {
    int k0 = 2 * i + half * 16;
    a[2 * i]     = rp[k0];
    a[2 * i + 1] = rp[k0 + 1];
  }
  return a;
}

// B operand (32x16) from [N][K] storage, elem (k,n)=base[n*ldb+k].
__device__ __forceinline__ v16bf load_b_frag_nk(const bf16_t* base, int ldb) {
  const int lane = threadIdx.x & 31;
  const int n    = lane & 15;
  const int half = lane >> 4;
  const bf16_t* cp = base + n * ldb + half * 16;
  v16bf b;
#pragma unroll
  for (int i = 0; i < 8; ++i) {
    b[2 * i]     = cp[2 * i];
    b[2 * i + 1] = cp[2 * i + 1];
  }
  return b;
}

// B operand (32x16) from [K][N] storage, elem (k,n)=base[k*ldb+n].
__device__ __forceinline__ v16bf load_b_frag_kn(const bf16_t* base, int ldb) {
  const int lane = threadIdx.x & 31;
  const int n    = lane & 15;
  const int half = lane >> 4;
  v16bf b;
#pragma unroll
  for (int i = 0; i < 8; ++i) {
    int k = 2 * i + half * 16;
    b[2 * i]     = base[k * ldb + n];
    b[2 * i + 1] = base[(k + 1) * ldb + n];
  }
  return b;
}

__device__ __forceinline__ v8f zero8() {
  v8f z;
#pragma unroll
  for (int i = 0; i < 8; ++i) z[i] = 0.f;
  return z;
}

__device__ __forceinline__ v8f wmma_bf16(v16bf a, v16bf b, v8f c) {
  return __builtin_amdgcn_wmma_f32_16x16x32_bf16(false, a, false, b, (short)0, c,
                                                 false, false);
}

// ---------------- Kernel 0: f32 -> bf16 conversion ----------------
__global__ void __launch_bounds__(256)
convert_f32_bf16(const float* __restrict__ src, bf16_t* __restrict__ dst, int n) {
  int i = blockIdx.x * 256 + threadIdx.x;
  if (i < n) dst[i] = (bf16_t)src[i];
}

// ---------------- Kernel 1: QKV projection ----------------
// Out[b,h,s,d] (bf16) = Xbf (8192x1024 bf16) @ W^T (bf16 [n][k]) + bias
// Block: 256 thr (8 waves), tile 128(M) x 64(N), K-step 32.
// A and B tiles double-buffered, staged by GLOBAL_LOAD_ASYNC_TO_LDS_B128.
__global__ void __launch_bounds__(256)
qkv_proj_kernel(const bf16_t* __restrict__ Xbf, const bf16_t* __restrict__ W,
                const float* __restrict__ bias, bf16_t* __restrict__ Out) {
  __shared__ bf16_t As[2][128][32];
  __shared__ bf16_t Bs[2][64][32];
  const int tid  = threadIdx.x;
  const int wave = tid >> 5;
  const int m0   = blockIdx.x * 128;
  const int n0   = blockIdx.y * 64;

  v8f acc[4];
#pragma unroll
  for (int t = 0; t < 4; ++t) acc[t] = zero8();

  // stage one k-step: A = 512 x b128, B = 256 x b128 (16B per thread-issue)
  auto stage = [&](int buf, int kt) {
    {
      int r = tid >> 1, seg = tid & 1;  // 128 rows x 2 segs (32B) -> first half
      async_b128(&As[buf][r][seg * 16],
                 &Xbf[(size_t)(m0 + r) * D_MODELC + kt + seg * 16]);
      async_b128(&As[buf][r][seg * 16 + 8],
                 &Xbf[(size_t)(m0 + r) * D_MODELC + kt + seg * 16 + 8]);
    }
    {
      int n = tid >> 2, seg = tid & 3;  // 64 rows x 4 segs of 16B
      async_b128(&Bs[buf][n][seg * 8],
                 &W[(size_t)(n0 + n) * D_MODELC + kt + seg * 8]);
    }
  };

  stage(0, 0);
  int cur = 0;
  for (int kt = 0; kt < D_MODELC; kt += 32) {
    wait_async0();
    __syncthreads();
    if (kt + 32 < D_MODELC) stage(cur ^ 1, kt + 32);

    v16bf a = load_a_frag(&As[cur][wave * 16][0], 32);
#pragma unroll
    for (int t = 0; t < 4; ++t) {
      v16bf b = load_b_frag_nk(&Bs[cur][t * 16][0], 32);
      acc[t] = wmma_bf16(a, b, acc[t]);
    }
    cur ^= 1;
  }

  const int lane = tid & 31;
  const int col = lane & 15;
  const int rowBase = (lane >> 4) * 8;
#pragma unroll
  for (int t = 0; t < 4; ++t) {
    int n = n0 + t * 16 + col;
    float bv = bias[n];
    int h = n >> 6, d = n & 63;
#pragma unroll
    for (int r = 0; r < 8; ++r) {
      int m = m0 + wave * 16 + rowBase + r;
      int bb = m >> 11, s = m & 2047;
      Out[(((size_t)bb * N_HEADC + h) * SEQC + s) * D_KC + d] =
          (bf16_t)(acc[t][r] + bv);
    }
  }
}

// ---------------- Kernel 2: attention scores ----------------
// attn[bh, q, k] = (Qh[bh,q,:] . Kh[bh,k,:]) / 8
__global__ void __launch_bounds__(256)
scores_kernel(const bf16_t* __restrict__ Qh, const bf16_t* __restrict__ Kh,
              float* __restrict__ attn) {
  __shared__ bf16_t Qs[16][64];
  const int tid  = threadIdx.x;
  const int wave = tid >> 5;
  const int q0   = blockIdx.x * 16;
  const int bh   = blockIdx.y;
  const bf16_t* Qbase = Qh + ((size_t)bh * SEQC + q0) * D_KC;

  if (tid < 128) {  // 16 rows x 8 segs of 16B, async DMA into LDS
    int r = tid >> 3, seg = tid & 7;
    async_b128(&Qs[r][seg * 8], &Qbase[r * D_KC + seg * 8]);
  }
  wait_async0();
  __syncthreads();

  v16bf a0 = load_a_frag(&Qs[0][0], 64);
  v16bf a1 = load_a_frag(&Qs[0][32], 64);

  const int lane = tid & 31;
  const int col = lane & 15;
  const int rowBase = (lane >> 4) * 8;

  for (int t = 0; t < 16; ++t) {
    int k0 = wave * 256 + t * 16;
    const bf16_t* Kbase = Kh + ((size_t)bh * SEQC + k0) * D_KC;
    v16bf b0 = load_b_frag_nk(Kbase, D_KC);
    v16bf b1 = load_b_frag_nk(Kbase + 32, D_KC);
    v8f acc = zero8();
    acc = wmma_bf16(a0, b0, acc);
    acc = wmma_bf16(a1, b1, acc);
    float* outp = attn + ((size_t)bh * SEQC + q0) * SEQC + k0;
#pragma unroll
    for (int r = 0; r < 8; ++r)
      outp[(size_t)(rowBase + r) * SEQC + col] = acc[r] * 0.125f;
  }
}

// ---------------- Kernel 3: row softmax over 2048, in place ----------------
__global__ void __launch_bounds__(256)
softmax_kernel(float* __restrict__ attn) {
  __shared__ float red[256];
  const int tid = threadIdx.x;
  float* row = attn + (size_t)blockIdx.x * SEQC;
  float v[8];
  float mx = -3.0e38f;
#pragma unroll
  for (int j = 0; j < 8; ++j) {
    v[j] = row[tid + j * 256];
    mx = fmaxf(mx, v[j]);
  }
  red[tid] = mx;
  __syncthreads();
  for (int s = 128; s > 0; s >>= 1) {
    if (tid < s) red[tid] = fmaxf(red[tid], red[tid + s]);
    __syncthreads();
  }
  mx = red[0];
  __syncthreads();
  float sum = 0.f;
#pragma unroll
  for (int j = 0; j < 8; ++j) {
    v[j] = __expf(v[j] - mx);
    sum += v[j];
  }
  red[tid] = sum;
  __syncthreads();
  for (int s = 128; s > 0; s >>= 1) {
    if (tid < s) red[tid] += red[tid + s];
    __syncthreads();
  }
  float inv = 1.f / red[0];
#pragma unroll
  for (int j = 0; j < 8; ++j) row[tid + j * 256] = v[j] * inv;
}

// ---------------- Kernel 4: ctx = attn @ Vh ----------------
// HBM-bound on the 1GB attn read; A needs f32->bf16 cvt so staged via VALU.
__global__ void __launch_bounds__(256)
ctx_kernel(const float* __restrict__ attn, const bf16_t* __restrict__ Vh,
           bf16_t* __restrict__ Ctx) {
  __shared__ bf16_t As[128][32];
  __shared__ bf16_t Bs[32][64];
  const int tid  = threadIdx.x;
  const int wave = tid >> 5;
  const int m0   = blockIdx.x * 128;
  const int bh   = blockIdx.y;
  const float*  Abase = attn + ((size_t)bh * SEQC + m0) * SEQC;
  const bf16_t* Vbase = Vh + (size_t)bh * SEQC * D_KC;

  v8f acc[4];
#pragma unroll
  for (int t = 0; t < 4; ++t) acc[t] = zero8();

  for (int kt = 0; kt < SEQC; kt += 32) {
    for (int e = tid; e < 128 * 32; e += 256) {
      int r = e >> 5, c = e & 31;
      As[r][c] = (bf16_t)Abase[(size_t)r * SEQC + kt + c];
    }
    {  // V tile 32x64 bf16 = 256 x b128, async DMA
      int k = tid >> 3, seg = tid & 7;
      async_b128(&Bs[k][seg * 8], &Vbase[(size_t)(kt + k) * D_KC + seg * 8]);
    }
    wait_async0();
    __syncthreads();
    v16bf a = load_a_frag(&As[wave * 16][0], 32);
#pragma unroll
    for (int t = 0; t < 4; ++t) {
      v16bf b = load_b_frag_kn(&Bs[0][t * 16], 64);
      acc[t] = wmma_bf16(a, b, acc[t]);
    }
    __syncthreads();
  }
  const int lane = tid & 31;
  const int col = lane & 15;
  const int rowBase = (lane >> 4) * 8;
  const int bb = bh >> 4, h = bh & 15;
#pragma unroll
  for (int t = 0; t < 4; ++t) {
#pragma unroll
    for (int r = 0; r < 8; ++r) {
      int s = m0 + wave * 16 + rowBase + r;
      Ctx[((size_t)bb * SEQC + s) * D_MODELC + h * D_KC + t * 16 + col] =
          (bf16_t)acc[t][r];
    }
  }
}

// ---------------- Kernel 5: out proj + residual + LayerNorm ----------------
// One block = 16 full rows. Wave w owns cols [w*128,+128). A tile async
// double-buffered; Wo fragments streamed from global (L2-resident, bf16).
__global__ void __launch_bounds__(256)
outproj_ln_kernel(const bf16_t* __restrict__ Ctx, const bf16_t* __restrict__ Wo,
                  const float* __restrict__ bo, const float* __restrict__ resid,
                  const float* __restrict__ gamma, const float* __restrict__ beta,
                  float* __restrict__ out) {
  __shared__ bf16_t As[2][16][32];
  __shared__ float sSum[16];
  __shared__ float sSq[16];
  const int tid  = threadIdx.x;
  const int wave = tid >> 5;
  const int m0   = blockIdx.x * 16;

  if (tid < 16) { sSum[tid] = 0.f; sSq[tid] = 0.f; }

  v8f acc[8];
#pragma unroll
  for (int t = 0; t < 8; ++t) acc[t] = zero8();

  auto stageA = [&](int buf, int kt) {
    if (tid < 64) {  // 16 rows x 4 segs of 16B
      int r = tid >> 2, seg = tid & 3;
      async_b128(&As[buf][r][seg * 8],
                 &Ctx[(size_t)(m0 + r) * D_MODELC + kt + seg * 8]);
    }
  };

  stageA(0, 0);
  int cur = 0;
  for (int kt = 0; kt < D_MODELC; kt += 32) {
    wait_async0();
    __syncthreads();
    if (kt + 32 < D_MODELC) stageA(cur ^ 1, kt + 32);

    v16bf a = load_a_frag(&As[cur][0][0], 32);
#pragma unroll
    for (int t = 0; t < 8; ++t) {
      const bf16_t* Wbase = Wo + (size_t)(wave * 128 + t * 16) * D_MODELC + kt;
      v16bf bfr = load_b_frag_nk(Wbase, D_MODELC);
      acc[t] = wmma_bf16(a, bfr, acc[t]);
    }
    cur ^= 1;
  }
  __syncthreads();

  const int lane = tid & 31;
  const int col = lane & 15;
  const int rowBase = (lane >> 4) * 8;
  float rs[8], rq[8];
#pragma unroll
  for (int r = 0; r < 8; ++r) { rs[r] = 0.f; rq[r] = 0.f; }

#pragma unroll
  for (int t = 0; t < 8; ++t) {
    int n = wave * 128 + t * 16 + col;
    float bv = bo[n];
#pragma unroll
    for (int r = 0; r < 8; ++r) {
      int m = m0 + rowBase + r;
      float val = acc[t][r] + bv + resid[(size_t)m * D_MODELC + n];
      acc[t][r] = val;
      rs[r] += val;
      rq[r] += val * val;
    }
  }
#pragma unroll
  for (int r = 0; r < 8; ++r) {
    atomicAdd(&sSum[rowBase + r], rs[r]);
    atomicAdd(&sSq[rowBase + r], rq[r]);
  }
  __syncthreads();

#pragma unroll
  for (int t = 0; t < 8; ++t) {
    int n = wave * 128 + t * 16 + col;
    float g = gamma[n], bt = beta[n];
#pragma unroll
    for (int r = 0; r < 8; ++r) {
      int m = m0 + rowBase + r;
      float mean = sSum[rowBase + r] * (1.f / 1024.f);
      float var  = sSq[rowBase + r] * (1.f / 1024.f) - mean * mean;
      float rstd = rsqrtf(var + LN_EPSC);
      out[(size_t)m * D_MODELC + n] = (acc[t][r] - mean) * rstd * g + bt;
    }
  }
}

// ---------------- Host launcher ----------------
extern "C" void kernel_launch(void* const* d_in, const int* in_sizes, int n_in,
                              void* d_out, int out_size, void* d_ws, size_t ws_size,
                              hipStream_t stream) {
  (void)in_sizes; (void)n_in; (void)out_size; (void)ws_size;
  const float* q     = (const float*)d_in[0];
  const float* k     = (const float*)d_in[1];
  const float* v     = (const float*)d_in[2];
  const float* Wq    = (const float*)d_in[3];
  const float* bq    = (const float*)d_in[4];
  const float* Wk    = (const float*)d_in[5];
  const float* bk    = (const float*)d_in[6];
  const float* Wv    = (const float*)d_in[7];
  const float* bv    = (const float*)d_in[8];
  const float* Wo    = (const float*)d_in[9];
  const float* bo    = (const float*)d_in[10];
  const float* gamma = (const float*)d_in[11];
  const float* beta  = (const float*)d_in[12];

  float* out  = (float*)d_out;
  float* attn = (float*)d_out + (size_t)ROWSC * D_MODELC;

  char* ws = (char*)d_ws;
  size_t off = 0;
  const size_t WBYTES = (size_t)D_MODELC * D_MODELC * sizeof(bf16_t);
  const size_t XBYTES = (size_t)ROWSC * D_MODELC * sizeof(bf16_t);
  const size_t HBYTES = (size_t)BATCHC * N_HEADC * SEQC * D_KC * sizeof(bf16_t);
  bf16_t* Wq_bf = (bf16_t*)(ws + off); off += WBYTES;
  bf16_t* Wk_bf = (bf16_t*)(ws + off); off += WBYTES;
  bf16_t* Wv_bf = (bf16_t*)(ws + off); off += WBYTES;
  bf16_t* Wo_bf = (bf16_t*)(ws + off); off += WBYTES;
  bf16_t* Xq_bf = (bf16_t*)(ws + off); off += XBYTES;
  bf16_t* Xk_bf = (bf16_t*)(ws + off); off += XBYTES;
  bf16_t* Xv_bf = (bf16_t*)(ws + off); off += XBYTES;
  bf16_t* Qh    = (bf16_t*)(ws + off); off += HBYTES;
  bf16_t* Kh    = (bf16_t*)(ws + off); off += HBYTES;
  bf16_t* Vh    = (bf16_t*)(ws + off); off += HBYTES;
  bf16_t* Ctx   = (bf16_t*)(ws + off); off += XBYTES;

  const int wn = D_MODELC * D_MODELC;
  convert_f32_bf16<<<wn / 256, 256, 0, stream>>>(Wq, Wq_bf, wn);
  convert_f32_bf16<<<wn / 256, 256, 0, stream>>>(Wk, Wk_bf, wn);
  convert_f32_bf16<<<wn / 256, 256, 0, stream>>>(Wv, Wv_bf, wn);
  convert_f32_bf16<<<wn / 256, 256, 0, stream>>>(Wo, Wo_bf, wn);
  const int xn = ROWSC * D_MODELC;
  convert_f32_bf16<<<xn / 256, 256, 0, stream>>>(q, Xq_bf, xn);
  convert_f32_bf16<<<xn / 256, 256, 0, stream>>>(k, Xk_bf, xn);
  convert_f32_bf16<<<xn / 256, 256, 0, stream>>>(v, Xv_bf, xn);

  dim3 gA(ROWSC / 128, D_MODELC / 64);  // 64 x 16
  qkv_proj_kernel<<<gA, 256, 0, stream>>>(Xq_bf, Wq_bf, bq, Qh);
  qkv_proj_kernel<<<gA, 256, 0, stream>>>(Xk_bf, Wk_bf, bk, Kh);
  qkv_proj_kernel<<<gA, 256, 0, stream>>>(Xv_bf, Wv_bf, bv, Vh);

  dim3 gS(SEQC / 16, BATCHC * N_HEADC);  // 128 x 64
  scores_kernel<<<gS, 256, 0, stream>>>(Qh, Kh, attn);

  softmax_kernel<<<BATCHC * N_HEADC * SEQC, 256, 0, stream>>>(attn);

  dim3 gC(SEQC / 128, BATCHC * N_HEADC);  // 16 x 64
  ctx_kernel<<<gC, 256, 0, stream>>>(attn, Vh, Ctx);

  outproj_ln_kernel<<<ROWSC / 16, 256, 0, stream>>>(Ctx, Wo_bf, bo, q, gamma,
                                                    beta, out);
}
